// RIMCell_83013127897912
// MI455X (gfx1250) — matrix operand order repaired and checked by reading, not compiled, not run on hardware
//
#include <hip/hip_runtime.h>
#include <hip/hip_bf16.h>

typedef __attribute__((ext_vector_type(16))) _Float16 v16h;
typedef __attribute__((ext_vector_type(8)))  float    v8f;

namespace {

constexpr int Bn      = 4096;
constexpr int NRIM_   = 6;
constexpr int UNITS_  = 600;
constexpr int INROW   = 518;          // BOUNDS + NRIM
constexpr int KP_X    = 512;          // K for x GEMMs
constexpr int KP_H    = 608;          // 600 padded to mult of 32
constexpr int KP_Z    = 1024;         // 400+600 padded
constexpr int KP_C    = 416;          // 400 padded
constexpr int NGATE   = 608;          // per-gate padded width (4*608 = 2432 total)
constexpr int NKVQ    = 656;          // 128 + 400 + 128
constexpr int NWOUT   = 608;          // 600 padded
constexpr int MW      = 4;            // waves (M-tiles) per GEMM block

// ---- WMMA operand layout helpers (gfx1250 wave32, 16x16x32 f16) ----------
// A (16x32): lane l -> M=l&15, group g=l>>4 ; element i -> K = kmapA(i,g)
// B (32x16): lane l -> N=l&15, group g=l>>4 ; element i -> K = i + 16*g
__device__ __forceinline__ int kmapA(int i, int g) {
  return i + 16 * g;
}
// inverse map: given row m (M dim) and k, flat index into swizzled A buffer
__device__ __forceinline__ size_t aswIdx(int KC, int m, int k) {
  int kc = k >> 5, r = k & 31;
  int i  = r & 15;
  int g  = r >> 4;
  int lane = g * 16 + (m & 15);
  return ((((size_t)(m >> 4)) * KC + kc) * 32 + lane) * 16 + i;
}

// ======================= weight / activation swizzles ======================

__global__ void swizzle_b_generic(const float* __restrict__ src,
                                  _Float16* __restrict__ dst,
                                  int Ksrc, int Nsrc, int Kp, int Np) {
  int z = blockIdx.z;
  size_t total = (size_t)Kp * Np;
  src += (size_t)z * Ksrc * Nsrc;
  dst += (size_t)z * total;
  size_t idx = (size_t)blockIdx.x * blockDim.x + threadIdx.x;
  if (idx >= total) return;
  int i = idx & 15, lane = (idx >> 4) & 31;
  size_t t = idx >> 9;
  int KC = Kp >> 5;
  int kc = (int)(t % KC), nt = (int)(t / KC);
  int g = lane >> 4;
  int n = nt * 16 + (lane & 15);
  int k = kc * 32 + i + 16 * g;
  float v = (k < Ksrc && n < Nsrc) ? src[(size_t)k * Nsrc + n] : 0.f;
  dst[idx] = (_Float16)v;
}

// head-mean of Wv folded into the B operand: (512 x 400)
__global__ void swizzle_b_wvm(const float* __restrict__ Wv, _Float16* __restrict__ dst) {
  size_t total = (size_t)KP_X * 400;
  size_t idx = (size_t)blockIdx.x * blockDim.x + threadIdx.x;
  if (idx >= total) return;
  int i = idx & 15, lane = (idx >> 4) & 31;
  size_t t = idx >> 9;
  int kc = (int)(t % 16), nt = (int)(t / 16);
  int g = lane >> 4;
  int n = nt * 16 + (lane & 15);
  int k = kc * 32 + i + 16 * g;
  const float* r = Wv + (size_t)k * 1600;
  dst[idx] = (_Float16)(0.25f * (r[n] + r[400 + n] + r[800 + n] + r[1200 + n]));
}

__global__ void bv_mean_kernel(const float* __restrict__ bv, float* __restrict__ bvm) {
  int j = blockIdx.x * blockDim.x + threadIdx.x;
  if (j < 400) bvm[j] = 0.25f * (bv[j] + bv[400 + j] + bv[800 + j] + bv[1200 + j]);
}

// concat [lstm_k ; lstm_r] -> (1024 x 4*608), gate-padded columns
__global__ void swizzle_b_wg(const float* __restrict__ lstm_k,
                             const float* __restrict__ lstm_r,
                             _Float16* __restrict__ dst) {
  int nr = blockIdx.z;
  const size_t total = (size_t)KP_Z * (4 * NGATE);
  size_t idx = (size_t)blockIdx.x * blockDim.x + threadIdx.x;
  if (idx >= total) return;
  int i = idx & 15, lane = (idx >> 4) & 31;
  size_t t = idx >> 9;
  const int KC = KP_Z >> 5;
  int kc = (int)(t % KC), nt = (int)(t / KC);
  int g = lane >> 4;
  int n = nt * 16 + (lane & 15);
  int k = kc * 32 + i + 16 * g;
  int gate = n / NGATE, u = n % NGATE;
  float v = 0.f;
  if (u < 600) {
    int c = gate * 600 + u;
    if (k < 400)       v = lstm_k[((size_t)nr * 400 + k) * 2400 + c];
    else if (k < 1000) v = lstm_r[((size_t)nr * 600 + (k - 400)) * 2400 + c];
  }
  dst[(size_t)nr * total + idx] = (_Float16)v;
}

// concat [Wkc | Wvc | Wqc] -> (608 x 656)
__global__ void swizzle_b_wkvq(const float* __restrict__ Wkc, const float* __restrict__ Wvc,
                               const float* __restrict__ Wqc, _Float16* __restrict__ dst) {
  int nr = blockIdx.z;
  const size_t total = (size_t)KP_H * NKVQ;
  size_t idx = (size_t)blockIdx.x * blockDim.x + threadIdx.x;
  if (idx >= total) return;
  int i = idx & 15, lane = (idx >> 4) & 31;
  size_t t = idx >> 9;
  const int KC = KP_H >> 5;
  int kc = (int)(t % KC), nt = (int)(t / KC);
  int g = lane >> 4;
  int n = nt * 16 + (lane & 15);
  int k = kc * 32 + i + 16 * g;
  float v = 0.f;
  if (k < 600) {
    size_t row = (size_t)nr * 600 + k;
    if (n < 128)      v = Wkc[row * 128 + n];
    else if (n < 528) v = Wvc[row * 400 + (n - 128)];
    else              v = Wqc[row * 128 + (n - 528)];
  }
  dst[(size_t)nr * total + idx] = (_Float16)v;
}

__global__ void swizzle_a_x(const float* __restrict__ in, _Float16* __restrict__ dst) {
  size_t total = (size_t)Bn * KP_X;
  size_t idx = (size_t)blockIdx.x * blockDim.x + threadIdx.x;
  if (idx >= total) return;
  int i = idx & 15, lane = (idx >> 4) & 31;
  size_t t = idx >> 9;
  int kc = (int)(t % 16), mt = (int)(t / 16);
  int g = lane >> 4;
  int m = mt * 16 + (lane & 15);
  int k = kc * 32 + kmapA(i, g);
  dst[idx] = (_Float16)in[(size_t)m * INROW + k];
}

__global__ void swizzle_a_hs(const float* __restrict__ hs, _Float16* __restrict__ dst) {
  int nr = blockIdx.z;
  const size_t total = (size_t)Bn * KP_H;
  size_t idx = (size_t)blockIdx.x * blockDim.x + threadIdx.x;
  if (idx >= total) return;
  int i = idx & 15, lane = (idx >> 4) & 31;
  size_t t = idx >> 9;
  const int KC = KP_H >> 5;
  int kc = (int)(t % KC), mt = (int)(t / KC);
  int g = lane >> 4;
  int m = mt * 16 + (lane & 15);
  int k = kc * 32 + kmapA(i, g);
  float v = (k < 600) ? hs[((size_t)m * NRIM_ + nr) * 600 + k] : 0.f;
  dst[(size_t)nr * total + idx] = (_Float16)v;
}

// hs half (+zero tail) of the fused gates input Z = [rnn_in(400) | hs(600) | 0(24)]
__global__ void swizzle_z_hs(const float* __restrict__ hs, _Float16* __restrict__ dst) {
  int nr = blockIdx.z;
  const size_t total = (size_t)Bn * KP_Z;
  size_t idx = (size_t)blockIdx.x * blockDim.x + threadIdx.x;
  if (idx >= total) return;
  int i = idx & 15, lane = (idx >> 4) & 31;
  size_t t = idx >> 9;
  const int KC = KP_Z >> 5;
  int kc = (int)(t % KC), mt = (int)(t / KC);
  int g = lane >> 4;
  int m = mt * 16 + (lane & 15);
  int k = kc * 32 + kmapA(i, g);
  if (k < 400) return;  // rnn_in section written by rnn_in_kernel
  float v = (k < 1000) ? hs[((size_t)m * NRIM_ + nr) * 600 + (k - 400)] : 0.f;
  dst[(size_t)nr * total + idx] = (_Float16)v;
}

// top-4-of-6 mask
__global__ void mask_kernel(const float* __restrict__ in, float* __restrict__ mask) {
  int b = blockIdx.x * blockDim.x + threadIdx.x;
  if (b >= Bn) return;
  float d[NRIM_];
#pragma unroll
  for (int n = 0; n < NRIM_; ++n) d[n] = in[(size_t)b * INROW + 512 + n];
#pragma unroll
  for (int n = 0; n < NRIM_; ++n) {
    int rank = 0;
#pragma unroll
    for (int m = 0; m < NRIM_; ++m)
      rank += (d[m] > d[n]) || (d[m] == d[n] && m < n);
    mask[(size_t)b * NRIM_ + n] = (rank < 4) ? 1.f : 0.f;
  }
}

// ============================== GEMM kernels ===============================

// Generic: C[m, n] = sum_k A[m,k]*B[k,n] (+ bias[n]).
// Block = MW waves; each wave owns one 16x16 tile (same N-tile, adjacent
// M-tiles, so the B/weight chunks are shared through the near cache).
__global__ void __launch_bounds__(32 * MW)
gemm_wmma(const _Float16* __restrict__ Asw,
          const _Float16* __restrict__ Bsw,
          const float* __restrict__ bias,
          float* __restrict__ C,
          int KC, int ldc,
          size_t aswZ, size_t bswZ, size_t cZ) {
  int nt = blockIdx.x;
  int mt = blockIdx.y * MW + threadIdx.y;
  int z  = blockIdx.z;
  int lane = threadIdx.x;
  const _Float16* A  = Asw + (size_t)z * aswZ + ((size_t)mt * KC * 32 + lane) * 16;
  const _Float16* Bp = Bsw + (size_t)z * bswZ + ((size_t)nt * KC * 32 + lane) * 16;
  v8f acc = {0.f, 0.f, 0.f, 0.f, 0.f, 0.f, 0.f, 0.f};
  for (int kc = 0; kc < KC; ++kc) {
    v16h a = *(const v16h*)(A + (size_t)kc * 512);
    v16h b = *(const v16h*)(Bp + (size_t)kc * 512);
    __builtin_prefetch(A + (size_t)(kc + 1) * 512, 0, 3);   // WGP-scope prefetch
    __builtin_prefetch(Bp + (size_t)(kc + 1) * 512, 0, 3);
    acc = __builtin_amdgcn_wmma_f32_16x16x32_f16(false, a, false, b, (short)0, acc,
                                                 false, false);
  }
  int n = nt * 16 + (lane & 15);
  int g = lane >> 4;
  float bb = bias ? bias[n] : 0.f;
  float* Cz = C + (size_t)z * cZ;
#pragma unroll
  for (int r = 0; r < 8; ++r) {
    int m = mt * 16 + r + 8 * g;
    Cz[(size_t)m * ldc + n] = acc[r] + bb;
  }
}

// scores2 + 2-way softmax + rnn_in, scattered into swizzled Z. 1 wave / (b,nr).
__global__ void rnn_in_kernel(const float* __restrict__ Qlay, const float* __restrict__ K0,
                              const float* __restrict__ bk,  const float* __restrict__ V20,
                              const float* __restrict__ bvm, const float* __restrict__ mask,
                              _Float16* __restrict__ Zsw) {
  int wid  = (int)((blockIdx.x * (size_t)blockDim.x + threadIdx.x) >> 5);
  int lane = threadIdx.x & 31;
  if (wid >= Bn * NRIM_) return;
  int b = wid / NRIM_, nr = wid % NRIM_;
  const float* q  = Qlay + ((size_t)b * NRIM_ + nr) * 256;
  const float* k0 = K0 + (size_t)b * 256;
  float s0 = 0.f, s1 = 0.f;
  for (int j = lane; j < 256; j += 32) {
    float qv = q[j];
    s0 += qv * k0[j];
    s1 += qv * bk[j];
  }
#pragma unroll
  for (int o = 16; o > 0; o >>= 1) {
    s0 += __shfl_down(s0, o, 32);
    s1 += __shfl_down(s1, o, 32);
  }
  s0 = __shfl(s0, 0, 32) * (1.f / 32.f);  // /4 heads /sqrt(64)
  s1 = __shfl(s1, 0, 32) * (1.f / 32.f);
  float mx = fmaxf(s0, s1);
  float e0 = __expf(s0 - mx), e1 = __expf(s1 - mx);
  float inv = 1.f / (e0 + e1);
  float p0 = e0 * inv, p1 = e1 * inv;
  float msk = mask[wid];
  _Float16* Z = Zsw + (size_t)nr * ((size_t)Bn * KP_Z);
  const int KC = KP_Z >> 5;
  for (int k = lane; k < 400; k += 32) {
    float v = (p0 * V20[(size_t)b * 400 + k] + p1 * bvm[k]) * msk;
    Z[aswIdx(KC, b, k)] = (_Float16)v;
  }
}

// gates GEMM (4 gate tiles per wave, A reused 4x) + fused LSTM epilogue
__global__ void __launch_bounds__(32 * MW)
gates_lstm_kernel(const _Float16* __restrict__ Zsw,
                  const _Float16* __restrict__ WgSw,
                  const float* __restrict__ lstm_b,
                  const float* __restrict__ cs,
                  const float* __restrict__ mask,
                  float* __restrict__ out_c,
                  float* __restrict__ out_h,
                  _Float16* __restrict__ HnSw) {
  int ut = blockIdx.x;
  int mt = blockIdx.y * MW + threadIdx.y;
  int nr = blockIdx.z;
  int lane = threadIdx.x;
  const int KC  = KP_Z >> 5;   // 32
  const int NTG = NGATE / 16;  // 38 tiles per gate
  const _Float16* A = Zsw + (size_t)nr * ((size_t)Bn * KP_Z) + ((size_t)mt * KC * 32 + lane) * 16;
  const _Float16* Wb = WgSw + (size_t)nr * ((size_t)KP_Z * 4 * NGATE);
  const _Float16* B0 = Wb + ((size_t)(0 * NTG + ut) * KC * 32 + lane) * 16;
  const _Float16* B1 = Wb + ((size_t)(1 * NTG + ut) * KC * 32 + lane) * 16;
  const _Float16* B2 = Wb + ((size_t)(2 * NTG + ut) * KC * 32 + lane) * 16;
  const _Float16* B3 = Wb + ((size_t)(3 * NTG + ut) * KC * 32 + lane) * 16;
  v8f aI = {0.f,0.f,0.f,0.f,0.f,0.f,0.f,0.f}, aF = aI, aG = aI, aO = aI;
  for (int kc = 0; kc < KC; ++kc) {
    v16h a  = *(const v16h*)(A  + (size_t)kc * 512);
    v16h b0 = *(const v16h*)(B0 + (size_t)kc * 512);
    v16h b1 = *(const v16h*)(B1 + (size_t)kc * 512);
    v16h b2 = *(const v16h*)(B2 + (size_t)kc * 512);
    v16h b3 = *(const v16h*)(B3 + (size_t)kc * 512);
    __builtin_prefetch(A + (size_t)(kc + 1) * 512, 0, 3);
    aI = __builtin_amdgcn_wmma_f32_16x16x32_f16(false, a, false, b0, (short)0, aI, false, false);
    aF = __builtin_amdgcn_wmma_f32_16x16x32_f16(false, a, false, b1, (short)0, aF, false, false);
    aG = __builtin_amdgcn_wmma_f32_16x16x32_f16(false, a, false, b2, (short)0, aG, false, false);
    aO = __builtin_amdgcn_wmma_f32_16x16x32_f16(false, a, false, b3, (short)0, aO, false, false);
  }
  int u = ut * 16 + (lane & 15);
  int g = lane >> 4;
  bool valid = (u < UNITS_);
  float bi = 0.f, bf = 0.f, bg = 0.f, bo = 0.f;
  if (valid) {
    const float* lb = lstm_b + (size_t)nr * 2400;
    bi = lb[u]; bf = lb[600 + u]; bg = lb[1200 + u]; bo = lb[1800 + u];
  }
  _Float16* Hn = HnSw + (size_t)nr * ((size_t)Bn * KP_H);
  const int KCH = KP_H >> 5;  // 19
#pragma unroll
  for (int r = 0; r < 8; ++r) {
    int b = mt * 16 + r + 8 * g;
    float hn = 0.f;
    if (valid) {
      size_t o = (size_t)b * (NRIM_ * UNITS_) + (size_t)nr * UNITS_ + u;
      float cp = cs[o];
      float si = 1.f / (1.f + __expf(-(aI[r] + bi)));
      float sf = 1.f / (1.f + __expf(-(aF[r] + bf)));
      float so = 1.f / (1.f + __expf(-(aO[r] + bo)));
      float tg = tanhf(aG[r] + bg);
      float c  = sf * cp + si * tg;
      float h  = so * tanhf(c);
      float msk = mask[(size_t)b * NRIM_ + nr];
      out_c[o] = (msk > 0.5f) ? c : cp;  // c_upd (blended)
      out_h[o] = h;                      // h_new (unblended; blended in hcomm)
      hn = h;
    }
    Hn[aswIdx(KCH, b, u)] = (_Float16)hn;  // tail u>=600 -> 0
  }
}

// per-(b, head, rim) comm attention; scatters ctx into swizzled layout
__global__ void comm_attn_kernel(const float* __restrict__ kvqc,
                                 const float* __restrict__ mask,
                                 _Float16* __restrict__ CtxSw) {
  int tid = (int)(blockIdx.x * (size_t)blockDim.x + threadIdx.x);
  if (tid >= Bn * 4 * NRIM_) return;
  int nr = tid % NRIM_;
  int h  = (tid / NRIM_) % 4;
  int b  = tid / (4 * NRIM_);
  const float* base = kvqc + (size_t)b * NRIM_ * NKVQ;
  const float* qv = base + (size_t)nr * NKVQ + 528 + h * 32;
  float sc[NRIM_], mx = -3.0e38f;
#pragma unroll
  for (int m = 0; m < NRIM_; ++m) {
    const float* kv = base + (size_t)m * NKVQ + h * 32;
    float s = 0.f;
#pragma unroll
    for (int d = 0; d < 32; ++d) s += qv[d] * kv[d];
    sc[m] = s * 0.17677669529663687f;  // 1/sqrt(32)
    mx = fmaxf(mx, sc[m]);
  }
  float sum = 0.f;
#pragma unroll
  for (int m = 0; m < NRIM_; ++m) { sc[m] = __expf(sc[m] - mx); sum += sc[m]; }
  float msk = mask[(size_t)b * NRIM_ + nr] / sum;
#pragma unroll
  for (int m = 0; m < NRIM_; ++m) sc[m] *= msk;
  _Float16* Ctx = CtxSw + (size_t)nr * ((size_t)Bn * KP_C);
  const int KC = KP_C >> 5;  // 13
  for (int d = 0; d < 100; ++d) {
    float v = 0.f;
#pragma unroll
    for (int m = 0; m < NRIM_; ++m) v += sc[m] * base[(size_t)m * NKVQ + 128 + h * 100 + d];
    Ctx[aswIdx(KC, b, h * 100 + d)] = (_Float16)v;
  }
  if (h == 0)  // zero the K padding tail (400..415)
    for (int k = 400; k < KP_C; ++k) Ctx[aswIdx(KC, b, k)] = (_Float16)0.f;
}

// ctx @ Wout + h_new, blended with hs by mask; writes h_upd and out_state
__global__ void __launch_bounds__(32 * MW)
hcomm_kernel(const _Float16* __restrict__ CtxSw,
             const _Float16* __restrict__ WoutSw,
             const float* __restrict__ hs,
             const float* __restrict__ mask,
             float* __restrict__ out_state,
             float* __restrict__ out_h) {
  int nt = blockIdx.x;
  int mt = blockIdx.y * MW + threadIdx.y;
  int nr = blockIdx.z;
  int lane = threadIdx.x;
  const int KC = KP_C >> 5;  // 13
  const _Float16* A  = CtxSw + (size_t)nr * ((size_t)Bn * KP_C) + ((size_t)mt * KC * 32 + lane) * 16;
  const _Float16* Bp = WoutSw + (size_t)nr * ((size_t)KP_C * NWOUT) + ((size_t)nt * KC * 32 + lane) * 16;
  v8f acc = {0.f,0.f,0.f,0.f,0.f,0.f,0.f,0.f};
  for (int kc = 0; kc < KC; ++kc) {
    v16h a = *(const v16h*)(A + (size_t)kc * 512);
    v16h b = *(const v16h*)(Bp + (size_t)kc * 512);
    __builtin_prefetch(A + (size_t)(kc + 1) * 512, 0, 3);
    acc = __builtin_amdgcn_wmma_f32_16x16x32_f16(false, a, false, b, (short)0, acc, false, false);
  }
  int u = nt * 16 + (lane & 15);
  int g = lane >> 4;
  if (u >= UNITS_) return;
#pragma unroll
  for (int r = 0; r < 8; ++r) {
    int b = mt * 16 + r + 8 * g;
    size_t o = (size_t)b * (NRIM_ * UNITS_) + (size_t)nr * UNITS_ + u;
    float msk = mask[(size_t)b * NRIM_ + nr];
    float v = (msk > 0.5f) ? (acc[r] + out_h[o]) : hs[o];
    out_h[o] = v;
    out_state[o] = v;
  }
}

}  // namespace

// ================================ launcher =================================

extern "C" void kernel_launch(void* const* d_in, const int* in_sizes, int n_in,
                              void* d_out, int out_size, void* d_ws, size_t ws_size,
                              hipStream_t stream) {
  const float* inputs = (const float*)d_in[0];
  const float* hs     = (const float*)d_in[1];
  const float* cs     = (const float*)d_in[2];
  const float* Wk     = (const float*)d_in[3];
  const float* bk     = (const float*)d_in[4];
  const float* Wv     = (const float*)d_in[5];
  const float* bv     = (const float*)d_in[6];
  const float* Wq     = (const float*)d_in[7];
  const float* lstm_k = (const float*)d_in[8];
  const float* lstm_r = (const float*)d_in[9];
  const float* lstm_b = (const float*)d_in[10];
  const float* Wkc    = (const float*)d_in[11];
  const float* Wvc    = (const float*)d_in[12];
  const float* Wqc    = (const float*)d_in[13];
  const float* Wout   = (const float*)d_in[14];

  float* out_state = (float*)d_out;
  float* out_h = out_state + (size_t)Bn * NRIM_ * UNITS_;
  float* out_c = out_h + (size_t)Bn * NRIM_ * UNITS_;

  char* ws = (char*)d_ws;
  size_t off = 0;
  auto take = [&](size_t bytes) -> char* {
    char* p = ws + off;
    off += (bytes + 255) & ~(size_t)255;
    return p;
  };
  _Float16* Xsw    = (_Float16*)take((size_t)Bn * KP_X * 2);
  _Float16* WkSw   = (_Float16*)take((size_t)KP_X * 256 * 2);
  _Float16* WvmSw  = (_Float16*)take((size_t)KP_X * 400 * 2);
  float*    bvm    = (float*)take(400 * 4);
  _Float16* WqSw   = (_Float16*)take((size_t)NRIM_ * KP_H * 256 * 2);
  _Float16* HsSw   = (_Float16*)take((size_t)NRIM_ * Bn * KP_H * 2);
  float*    K0     = (float*)take((size_t)Bn * 256 * 4);
  float*    V20    = (float*)take((size_t)Bn * 400 * 4);
  float*    Qlay   = (float*)take((size_t)Bn * NRIM_ * 256 * 4);
  float*    maskb  = (float*)take((size_t)Bn * NRIM_ * 4);
  _Float16* Zsw    = (_Float16*)take((size_t)NRIM_ * Bn * KP_Z * 2);
  _Float16* WgSw   = (_Float16*)take((size_t)NRIM_ * KP_Z * 4 * NGATE * 2);
  _Float16* HnSw   = (_Float16*)take((size_t)NRIM_ * Bn * KP_H * 2);
  _Float16* WkvqSw = (_Float16*)take((size_t)NRIM_ * KP_H * NKVQ * 2);
  float*    kvqc   = (float*)take((size_t)Bn * NRIM_ * NKVQ * 4);
  _Float16* CtxSw  = (_Float16*)take((size_t)NRIM_ * Bn * KP_C * 2);
  _Float16* WoutSw = (_Float16*)take((size_t)NRIM_ * KP_C * NWOUT * 2);

  auto blks = [](size_t total) { return (unsigned)((total + 255) / 256); };
  const dim3 gemmBlk(32, MW, 1);
  const unsigned MTB = Bn / 16 / MW;  // 64 M-tile blocks

  // ---- operand preparation (runs every call; deterministic) ----
  swizzle_b_generic<<<dim3(blks((size_t)KP_X * 256), 1, 1), 256, 0, stream>>>(
      Wk, WkSw, 512, 256, KP_X, 256);
  swizzle_b_wvm<<<dim3(blks((size_t)KP_X * 400), 1, 1), 256, 0, stream>>>(Wv, WvmSw);
  bv_mean_kernel<<<dim3(2), 256, 0, stream>>>(bv, bvm);
  swizzle_b_generic<<<dim3(blks((size_t)KP_H * 256), 1, NRIM_), 256, 0, stream>>>(
      Wq, WqSw, 600, 256, KP_H, 256);
  swizzle_b_wg<<<dim3(blks((size_t)KP_Z * 4 * NGATE), 1, NRIM_), 256, 0, stream>>>(
      lstm_k, lstm_r, WgSw);
  swizzle_b_wkvq<<<dim3(blks((size_t)KP_H * NKVQ), 1, NRIM_), 256, 0, stream>>>(
      Wkc, Wvc, Wqc, WkvqSw);
  swizzle_b_generic<<<dim3(blks((size_t)KP_C * NWOUT), 1, NRIM_), 256, 0, stream>>>(
      Wout, WoutSw, 400, 600, KP_C, NWOUT);
  swizzle_a_x<<<dim3(blks((size_t)Bn * KP_X), 1, 1), 256, 0, stream>>>(inputs, Xsw);
  swizzle_a_hs<<<dim3(blks((size_t)Bn * KP_H), 1, NRIM_), 256, 0, stream>>>(hs, HsSw);
  swizzle_z_hs<<<dim3(blks((size_t)Bn * KP_Z), 1, NRIM_), 256, 0, stream>>>(hs, Zsw);
  mask_kernel<<<dim3(Bn / 256), 256, 0, stream>>>(inputs, maskb);

  // ---- input attention: k0 = x@Wk+bk ; v2_0 = x@mean_h(Wv)+mean_h(bv) ; q ----
  gemm_wmma<<<dim3(256 / 16, MTB, 1), gemmBlk, 0, stream>>>(
      Xsw, WkSw, bk, K0, KP_X / 32, 256, 0, 0, 0);
  gemm_wmma<<<dim3(400 / 16, MTB, 1), gemmBlk, 0, stream>>>(
      Xsw, WvmSw, bvm, V20, KP_X / 32, 400, 0, 0, 0);
  gemm_wmma<<<dim3(256 / 16, MTB, NRIM_), gemmBlk, 0, stream>>>(
      HsSw, WqSw, nullptr, Qlay, KP_H / 32, NRIM_ * 256,
      (size_t)Bn * KP_H, (size_t)KP_H * 256, 256);

  // ---- scores2 softmax + rnn_in (fills the rest of Z) ----
  rnn_in_kernel<<<dim3((Bn * NRIM_ * 32) / 256), 256, 0, stream>>>(
      Qlay, K0, bk, V20, bvm, maskb, Zsw);

  // ---- gates GEMM + fused LSTM ----
  gates_lstm_kernel<<<dim3(NGATE / 16, MTB, NRIM_), gemmBlk, 0, stream>>>(
      Zsw, WgSw, lstm_b, cs, maskb, out_c, out_h, HnSw);

  // ---- comm projections kc|vc|qc ----
  gemm_wmma<<<dim3(NKVQ / 16, MTB, NRIM_), gemmBlk, 0, stream>>>(
      HnSw, WkvqSw, nullptr, kvqc, KP_H / 32, NRIM_ * NKVQ,
      (size_t)Bn * KP_H, (size_t)KP_H * NKVQ, NKVQ);

  // ---- comm attention ----
  comm_attn_kernel<<<dim3(blks((size_t)Bn * 4 * NRIM_)), 256, 0, stream>>>(
      kvqc, maskb, CtxSw);

  // ---- h_comm = ctx@Wout + h_new, blended; writes out_state & h_upd ----
  hcomm_kernel<<<dim3(NWOUT / 16, MTB, NRIM_), gemmBlk, 0, stream>>>(
      CtxSw, WoutSw, hs, maskb, out_state, out_h);

  (void)in_sizes; (void)n_in; (void)out_size; (void)ws_size;
}
